// AdaptiveFourierNeuralOperator_8443905703994
// MI455X (gfx1250) — matrix operand order repaired and checked by reading, not compiled, not run on hardware
//
#include <hip/hip_runtime.h>

// ---------------------------------------------------------------------------
// AFNO block for MI455X (gfx1250, wave32, WMMA).
// rfft / complex layers / irfft all run as bf16 WMMA GEMMs; the rank-4
// MoE + gate runs on the VALU.  Intermediate spectra live in LDS as bf16
// pre-swizzled into WMMA A-fragment layout (2x ds_load_b128 per fragment).
// B-fragments are software double-buffered so LDS latency overlaps the
// matrix pipe; outer tile loops stay rolled to keep VGPRs < 256.
// ---------------------------------------------------------------------------

typedef __attribute__((ext_vector_type(16))) __bf16 v16bf;
typedef __attribute__((ext_vector_type(8)))  __bf16 v8bf;
typedef __attribute__((ext_vector_type(8)))  float  v8f;

#define F_BINS 129
#define FP     160          // F padded: 10 column tiles / 5 K tiles
#define CCH    256
#define NN     2048
#define BB     32
#define NT_RF  10           // FP/16 output tiles (rfft, layers)
#define KT_RF  8            // 256/32 K tiles over c  (rfft)
#define KT_L   5            // FP/32 K tiles over f   (layers, irfft)
#define NT_IR  16           // 256/16 output tiles    (irfft)

// workspace layout (bytes)
#define SZ_CRCI  (2 * CCH * FP * 2)     // 163840  rfft  R+I, bf16, frag-swizzled
#define SZ_W     (2 * FP * FP * 2)      // 102400  w R+I
#define SZ_DRDI  (2 * FP * CCH * 2)     // 163840  irfft R+I
#define OFF_CRCI 0
#define OFF_W1   (OFF_CRCI + SZ_CRCI)
#define OFF_W2   (OFF_W1 + SZ_W)
#define OFF_DRDI (OFF_W2 + SZ_W)
#define OFF_F32  (OFF_DRDI + SZ_DRDI)   // 640 padded bias floats + 128 att floats

// dynamic LDS layout
#define LMAT_BYTES 163840               // largest phase matrix blob
#define BXC_BYTES  (4 * 2 * 16 * FP * 2)// 4 waves * (Xr,Xi) 16x160 bf16 = 40960
#define CBUF_BYTES (4 * 16 * 4 * 2 * 4) // 4 waves * 16 rows * 4 g * (re,im)
#define SMEM_BYTES (LMAT_BYTES + BXC_BYTES + CBUF_BYTES)

static __device__ __forceinline__ v8f zero8() {
    v8f z;
#pragma unroll
    for (int i = 0; i < 8; ++i) z[i] = 0.0f;
    return z;
}

static __device__ __forceinline__ v8f wmma_bf16(v16bf a, v16bf b, v8f c) {
    return __builtin_amdgcn_wmma_f32_16x16x32_bf16(false, a, false, b,
                                                   (short)0, c, false, false);
}

// Store one fp32 value into a bf16 buffer laid out in WMMA A-fragment order:
// fragment kt: lane L holds 16 elems; elem e <-> k = kt*32 + (L>>4)*8 + (e&7)
// + (e>>3)*16, row M = L&15.  Inverse mapping used here.
static __device__ __forceinline__ void store_swz(__bf16* buf, int M, int f, float v) {
    int kt = f >> 5, kl = f & 31;
    int lane_a = (((kl >> 3) & 1) << 4) | M;
    int e = (kl & 7) | ((kl >> 4) << 3);
    buf[(kt * 32 + lane_a) * 16 + e] = (__bf16)v;
}

static __device__ __forceinline__ void lds_copy16(char* dst, const char* src,
                                                  int nbytes, int tid, int nthr) {
    uint4* d = (uint4*)dst;
    const uint4* s = (const uint4*)src;
    int n = nbytes >> 4;
    for (int i = tid; i < n; i += nthr) d[i] = s[i];
}

static __device__ __forceinline__ void prefetch_blob(const char* p, int nbytes,
                                                     int tid, int nthr) {
    for (int i = tid * 256; i < nbytes; i += nthr * 256)
        __builtin_prefetch(p + i, 0, 3);           // -> global_prefetch_b8
}

// ---------------------------------------------------------------------------
// setup: fragment-swizzled matrices (B-operand layout).  Blob element t:
//   isI = t >= half;  frag = u/512; e = u%512; lane = e/16; j = e%16
//   k = kt*32 + (lane>>4)*8 + (j&7) + (j>>3)*16 ; n = ft*16 + (lane&15)
// ---------------------------------------------------------------------------
__global__ void build_rfft_mat(__bf16* __restrict__ blob) {
    int t = blockIdx.x * blockDim.x + threadIdx.x;
    const int half_sz = CCH * FP;
    if (t >= 2 * half_sz) return;
    int isI = t >= half_sz;
    int u = t - isI * half_sz;
    int frag = u >> 9, e = u & 511, lane = e >> 4, j = e & 15;
    int kt = frag / NT_RF, ft = frag % NT_RF;
    int c = kt * 32 + ((lane >> 4) << 3) + (j & 7) + ((j >> 3) << 4);     // K index = channel
    int f = ft * 16 + (lane & 15);                // N index = freq bin
    float v = 0.0f;
    if (f < F_BINS) {
        float ang = 6.283185307179586f * (float)((c * f) & 255) * (1.0f / 256.0f);
        v = (isI ? -__sinf(ang) : __cosf(ang)) * 0.0625f;   // ortho 1/16
    }
    blob[t] = (__bf16)v;
}

__global__ void build_irfft_mat(__bf16* __restrict__ blob) {
    int t = blockIdx.x * blockDim.x + threadIdx.x;
    const int half_sz = FP * CCH;
    if (t >= 2 * half_sz) return;
    int isI = t >= half_sz;
    int u = t - isI * half_sz;
    int frag = u >> 9, e = u & 511, lane = e >> 4, j = e & 15;
    int kt = frag / NT_IR, ct = frag % NT_IR;
    int f = kt * 32 + ((lane >> 4) << 3) + (j & 7) + ((j >> 3) << 4);     // K index = freq bin
    int c = ct * 16 + (lane & 15);                // N index = channel
    float v = 0.0f;
    if (f < F_BINS) {
        float wgt = (f == 0 || f == 128) ? 1.0f : 2.0f;  // hermitian fold
        float ang = 6.283185307179586f * (float)((f * c) & 255) * (1.0f / 256.0f);
        v = (isI ? -__sinf(ang) : __cosf(ang)) * wgt * 0.0625f;
    }
    blob[t] = (__bf16)v;
}

__global__ void build_w_mat(const float* __restrict__ w,   // [2][129][129]
                            __bf16* __restrict__ blob) {
    int t = blockIdx.x * blockDim.x + threadIdx.x;
    const int half_sz = FP * FP;
    if (t >= 2 * half_sz) return;
    int isI = t >= half_sz;
    int u = t - isI * half_sz;
    int frag = u >> 9, e = u & 511, lane = e >> 4, j = e & 15;
    int kt = frag / NT_RF, ft = frag % NT_RF;
    int fin = kt * 32 + ((lane >> 4) << 3) + (j & 7) + ((j >> 3) << 4);
    int fout = ft * 16 + (lane & 15);
    float v = 0.0f;
    if (fin < F_BINS && fout < F_BINS)
        v = w[isI * F_BINS * F_BINS + fin * F_BINS + fout];
    blob[t] = (__bf16)v;
}

// gate + padded biases.  fsec: [b1r 160][b1i 160][b2r 160][b2i 160][att 4x32]
__global__ void build_gate(const float* __restrict__ time_gra,
                           const float* __restrict__ emb_w,
                           const float* __restrict__ emb_b,
                           const float* __restrict__ gra_feature,
                           const float* __restrict__ b1,
                           const float* __restrict__ b2,
                           float* __restrict__ fsec) {
    int t = threadIdx.x;
    for (int i = t; i < 4 * FP; i += blockDim.x) {
        int which = i / FP, f = i % FP;
        const float* src = (which < 2) ? b1 : b2;
        fsec[i] = (f < F_BINS) ? src[(which & 1) * F_BINS + f] : 0.0f;
    }
    if (t < BB) {
        int b = t;
        float tg[5], tsum = 0.0f;
#pragma unroll
        for (int k = 0; k < 5; ++k) { tg[k] = time_gra[b * 5 + k]; tsum += tg[k]; }
        float lg[4] = {0.f, 0.f, 0.f, 0.f};
        for (int f = 0; f < F_BINS; ++f) {
            float gr = emb_b[f];
#pragma unroll
            for (int k = 0; k < 5; ++k) gr += tg[k] * emb_w[k * F_BINS + f];
#pragma unroll
            for (int g = 0; g < 4; ++g) lg[g] += gr * gra_feature[g * F_BINS + f];
        }
        float mx = fmaxf(fmaxf(lg[0], lg[1]), fmaxf(lg[2], lg[3]));
        float e[4], s = 0.0f;
#pragma unroll
        for (int g = 0; g < 4; ++g) { e[g] = __expf(lg[g] - mx); s += e[g]; }
        float nm = (tsum == 0.0f) ? 0.0f : 1.0f;   // null-batch mask
#pragma unroll
        for (int g = 0; g < 4; ++g)
            fsec[4 * FP + g * BB + b] = 4.0f * (e[g] / s) * nm;  // LORA_SCALE folded
    }
}

// ---------------------------------------------------------------------------
// main fused kernel: 4 waves/WG, each wave owns 16 n-rows of one batch.
// ---------------------------------------------------------------------------
__global__ __launch_bounds__(128) void afno_main(
    const float* __restrict__ x, const __bf16* __restrict__ crci,
    const __bf16* __restrict__ w1b, const __bf16* __restrict__ w2b,
    const __bf16* __restrict__ drdi, const float* __restrict__ fsec,
    const float* __restrict__ A1r, const float* __restrict__ A1i,
    const float* __restrict__ B1r, const float* __restrict__ B1i,
    const float* __restrict__ A2r, const float* __restrict__ A2i,
    const float* __restrict__ B2r, const float* __restrict__ B2i,
    float* __restrict__ out) {
    extern __shared__ char smem[];
    __bf16* lmat = (__bf16*)smem;

    const int tid = threadIdx.x, lane = tid & 31, wave = tid >> 5;
    const int wg = blockIdx.x;
    const int bidx = wg >> 5;                  // batch
    const int n0 = (wg & 31) * 64 + wave * 16; // first n-row of this wave
    const int ln = lane & 15, half = lane >> 4;

    // per-wave bf16 spectra, A-fragment-swizzled (Xr then Xi)
    __bf16* bxr = (__bf16*)(smem + LMAT_BYTES) + wave * (2 * 16 * FP);
    __bf16* bxi = bxr + 16 * FP;
    float* cbuf = (float*)(smem + LMAT_BYTES + BXC_BYTES) + wave * 128;
    const float* att = fsec + 4 * FP;

    // ---- phase 0: rfft ---------------------------------------------------
    lds_copy16(smem, (const char*)crci, SZ_CRCI, tid, blockDim.x);
    __syncthreads();
    prefetch_blob((const char*)w1b, SZ_W, tid, blockDim.x);
    {
        const float* xrow = x + ((size_t)bidx * NN + n0) * CCH;
        v16bf XA[KT_RF];
#pragma unroll
        for (int kt = 0; kt < KT_RF; ++kt) {
            v16bf a;
#pragma unroll
            for (int e = 0; e < 16; e += 2) {   // (e,e+1) -> contiguous k pair
                int k = kt * 32 + half * 8 + (e & 7) + ((e >> 3) << 4);
                float2 p = *(const float2*)(xrow + ln * CCH + k);
                a[e] = (__bf16)p.x;
                a[e + 1] = (__bf16)p.y;
            }
            XA[kt] = a;
        }
        const v16bf* mR = (const v16bf*)lmat;
        const v16bf* mI = (const v16bf*)(lmat + CCH * FP);
#pragma unroll 1                               // keep one tile's accs live
        for (int ft = 0; ft < NT_RF; ++ft) {
            v8f aR = zero8(), aI = zero8();
            v16bf bR = mR[ft * 32 + lane];     // kt = 0 fragments
            v16bf bI = mI[ft * 32 + lane];
#pragma unroll
            for (int kt = 0; kt < KT_RF; ++kt) {
                v16bf bRn, bIn;
                if (kt + 1 < KT_RF) {          // double-buffer next fragments
                    int fr = (kt + 1) * NT_RF + ft;
                    bRn = mR[fr * 32 + lane];
                    bIn = mI[fr * 32 + lane];
                }
                aR = wmma_bf16(XA[kt], bR, aR);
                aI = wmma_bf16(XA[kt], bI, aI);
                bR = bRn;
                bI = bIn;
            }
            int f = ft * 16 + ln;
#pragma unroll
            for (int r = 0; r < 8; ++r) {
                int row = r + 8 * half;
                store_swz(bxr, row, f, aR[r]);
                store_swz(bxi, row, f, aI[r]);
            }
        }
    }
    __syncthreads();

    // ---- phase 1: stage W1, MoE-1 dots, layer-1 A fragments --------------
    lds_copy16(smem, (const char*)w1b, SZ_W, tid, blockDim.x);
    {   // complex dots vs expert B-vectors: lanes 0-15 -> g{0,1}, 16-31 -> g{2,3}
        int row = ln, g0 = half * 2;
        float ur[2] = {0.f, 0.f}, ui[2] = {0.f, 0.f};
        for (int m = 0; m < 16; ++m) {          // f = 8m..8m+7, contiguous in swz
            int kt = m >> 2, lane_a = ((m & 1) << 4) | row, e0 = ((m >> 1) & 1) << 3;
            int idx = (kt * 32 + lane_a) * 16 + e0;
            v8bf xrv = *(const v8bf*)(bxr + idx);
            v8bf xiv = *(const v8bf*)(bxi + idx);
#pragma unroll
            for (int j = 0; j < 8; ++j) {
                int f = m * 8 + j;
                float xr = (float)xrv[j], xi = (float)xiv[j];
#pragma unroll
                for (int tg = 0; tg < 2; ++tg) {
                    int g = g0 + tg;
                    float br = B1r[g * F_BINS + f], bi = B1i[g * F_BINS + f];
                    ur[tg] += xr * br - xi * bi;
                    ui[tg] += xr * bi + xi * br;
                }
            }
        }
        {   // f = 128 tail
            int idx = (4 * 32 + row) * 16;
            float xr = (float)bxr[idx], xi = (float)bxi[idx];
#pragma unroll
            for (int tg = 0; tg < 2; ++tg) {
                int g = g0 + tg;
                float br = B1r[g * F_BINS + 128], bi = B1i[g * F_BINS + 128];
                ur[tg] += xr * br - xi * bi;
                ui[tg] += xr * bi + xi * br;
            }
        }
#pragma unroll
        for (int tg = 0; tg < 2; ++tg) {
            int g = g0 + tg;
            float a = att[g * BB + bidx];
            cbuf[row * 8 + g * 2 + 0] = a * ur[tg];
            cbuf[row * 8 + g * 2 + 1] = a * ui[tg];
        }
    }
    v16bf AR[KT_L], AI[KT_L];
#pragma unroll
    for (int kt = 0; kt < KT_L; ++kt) {
        AR[kt] = ((const v16bf*)bxr)[kt * 32 + lane];
        AI[kt] = ((const v16bf*)bxi)[kt * 32 + lane];
    }
    __syncthreads();

    // ---- phase 2: layer 1 (relu) + MoE-1 ---------------------------------
    prefetch_blob((const char*)w2b, SZ_W, tid, blockDim.x);
    {
        const v16bf* wR = (const v16bf*)lmat;
        const v16bf* wI = (const v16bf*)(lmat + FP * FP);
#pragma unroll 1
        for (int ft = 0; ft < NT_RF; ++ft) {
            v8f aRR = zero8(), aII = zero8(), aRI = zero8(), aIR = zero8();
            v16bf bR = wR[ft * 32 + lane];
            v16bf bI = wI[ft * 32 + lane];
#pragma unroll
            for (int kt = 0; kt < KT_L; ++kt) {
                v16bf bRn, bIn;
                if (kt + 1 < KT_L) {
                    int fr = (kt + 1) * NT_RF + ft;
                    bRn = wR[fr * 32 + lane];
                    bIn = wI[fr * 32 + lane];
                }
                aRR = wmma_bf16(AR[kt], bR, aRR);
                aII = wmma_bf16(AI[kt], bI, aII);
                aRI = wmma_bf16(AR[kt], bI, aRI);
                aIR = wmma_bf16(AI[kt], bR, aIR);
                bR = bRn;
                bI = bIn;
            }
            int f = ft * 16 + ln;
            float br = fsec[f], bi = fsec[FP + f];
            float avr[4], avi[4];
            bool inb = f < F_BINS;
            int fi = inb ? f : 0;
#pragma unroll
            for (int g = 0; g < 4; ++g) {
                avr[g] = inb ? A1r[g * F_BINS + fi] : 0.0f;
                avi[g] = inb ? A1i[g * F_BINS + fi] : 0.0f;
            }
#pragma unroll
            for (int r = 0; r < 8; ++r) {
                int row = r + 8 * half;
                float yr = fmaxf(aRR[r] - aII[r] + br, 0.0f);
                float yi = fmaxf(aRI[r] + aIR[r] + bi, 0.0f);
#pragma unroll
                for (int g = 0; g < 4; ++g) {
                    float cr = cbuf[row * 8 + g * 2 + 0];
                    float ci = cbuf[row * 8 + g * 2 + 1];
                    yr += cr * avr[g] - ci * avi[g];
                    yi += cr * avi[g] + ci * avr[g];
                }
                store_swz(bxr, row, f, yr);
                store_swz(bxi, row, f, yi);
            }
        }
    }
    __syncthreads();

    // ---- phase 3: stage W2, MoE-2 dots, layer-2 A fragments --------------
    lds_copy16(smem, (const char*)w2b, SZ_W, tid, blockDim.x);
    {
        int row = ln, g0 = half * 2;
        float ur[2] = {0.f, 0.f}, ui[2] = {0.f, 0.f};
        for (int m = 0; m < 16; ++m) {
            int kt = m >> 2, lane_a = ((m & 1) << 4) | row, e0 = ((m >> 1) & 1) << 3;
            int idx = (kt * 32 + lane_a) * 16 + e0;
            v8bf xrv = *(const v8bf*)(bxr + idx);
            v8bf xiv = *(const v8bf*)(bxi + idx);
#pragma unroll
            for (int j = 0; j < 8; ++j) {
                int f = m * 8 + j;
                float xr = (float)xrv[j], xi = (float)xiv[j];
#pragma unroll
                for (int tg = 0; tg < 2; ++tg) {
                    int g = g0 + tg;
                    float br = B2r[g * F_BINS + f], bi = B2i[g * F_BINS + f];
                    ur[tg] += xr * br - xi * bi;
                    ui[tg] += xr * bi + xi * br;
                }
            }
        }
        {
            int idx = (4 * 32 + row) * 16;
            float xr = (float)bxr[idx], xi = (float)bxi[idx];
#pragma unroll
            for (int tg = 0; tg < 2; ++tg) {
                int g = g0 + tg;
                float br = B2r[g * F_BINS + 128], bi = B2i[g * F_BINS + 128];
                ur[tg] += xr * br - xi * bi;
                ui[tg] += xr * bi + xi * br;
            }
        }
#pragma unroll
        for (int tg = 0; tg < 2; ++tg) {
            int g = g0 + tg;
            float a = att[g * BB + bidx];
            cbuf[row * 8 + g * 2 + 0] = a * ur[tg];
            cbuf[row * 8 + g * 2 + 1] = a * ui[tg];
        }
    }
#pragma unroll
    for (int kt = 0; kt < KT_L; ++kt) {
        AR[kt] = ((const v16bf*)bxr)[kt * 32 + lane];
        AI[kt] = ((const v16bf*)bxi)[kt * 32 + lane];
    }
    __syncthreads();

    // ---- phase 4: layer 2 (no relu) + MoE-2 ------------------------------
    prefetch_blob((const char*)drdi, SZ_DRDI, tid, blockDim.x);
    {
        const v16bf* wR = (const v16bf*)lmat;
        const v16bf* wI = (const v16bf*)(lmat + FP * FP);
#pragma unroll 1
        for (int ft = 0; ft < NT_RF; ++ft) {
            v8f aRR = zero8(), aII = zero8(), aRI = zero8(), aIR = zero8();
            v16bf bR = wR[ft * 32 + lane];
            v16bf bI = wI[ft * 32 + lane];
#pragma unroll
            for (int kt = 0; kt < KT_L; ++kt) {
                v16bf bRn, bIn;
                if (kt + 1 < KT_L) {
                    int fr = (kt + 1) * NT_RF + ft;
                    bRn = wR[fr * 32 + lane];
                    bIn = wI[fr * 32 + lane];
                }
                aRR = wmma_bf16(AR[kt], bR, aRR);
                aII = wmma_bf16(AI[kt], bI, aII);
                aRI = wmma_bf16(AR[kt], bI, aRI);
                aIR = wmma_bf16(AI[kt], bR, aIR);
                bR = bRn;
                bI = bIn;
            }
            int f = ft * 16 + ln;
            float br = fsec[2 * FP + f], bi = fsec[3 * FP + f];
            float avr[4], avi[4];
            bool inb = f < F_BINS;
            int fi = inb ? f : 0;
#pragma unroll
            for (int g = 0; g < 4; ++g) {
                avr[g] = inb ? A2r[g * F_BINS + fi] : 0.0f;
                avi[g] = inb ? A2i[g * F_BINS + fi] : 0.0f;
            }
#pragma unroll
            for (int r = 0; r < 8; ++r) {
                int row = r + 8 * half;
                float yr = aRR[r] - aII[r] + br;
                float yi = aRI[r] + aIR[r] + bi;
#pragma unroll
                for (int g = 0; g < 4; ++g) {
                    float cr = cbuf[row * 8 + g * 2 + 0];
                    float ci = cbuf[row * 8 + g * 2 + 1];
                    yr += cr * avr[g] - ci * avi[g];
                    yi += cr * avi[g] + ci * avr[g];
                }
                store_swz(bxr, row, f, yr);
                store_swz(bxi, row, f, yi);
            }
        }
    }
    __syncthreads();

    // ---- phase 5: irfft ---------------------------------------------------
    lds_copy16(smem, (const char*)drdi, SZ_DRDI, tid, blockDim.x);
#pragma unroll
    for (int kt = 0; kt < KT_L; ++kt) {
        AR[kt] = ((const v16bf*)bxr)[kt * 32 + lane];
        AI[kt] = ((const v16bf*)bxi)[kt * 32 + lane];
    }
    __syncthreads();
    {
        const v16bf* dR = (const v16bf*)lmat;
        const v16bf* dI = (const v16bf*)(lmat + FP * CCH);
#pragma unroll 1
        for (int ct = 0; ct < NT_IR; ++ct) {
            v8f acc = zero8();
            v16bf bR = dR[ct * 32 + lane];
            v16bf bI = dI[ct * 32 + lane];
#pragma unroll
            for (int kt = 0; kt < KT_L; ++kt) {
                v16bf bRn, bIn;
                if (kt + 1 < KT_L) {
                    int fr = (kt + 1) * NT_IR + ct;
                    bRn = dR[fr * 32 + lane];
                    bIn = dI[fr * 32 + lane];
                }
                acc = wmma_bf16(AR[kt], bR, acc);
                acc = wmma_bf16(AI[kt], bI, acc);
                bR = bRn;
                bI = bIn;
            }
#pragma unroll
            for (int r = 0; r < 8; ++r) {
                size_t o = ((size_t)bidx * NN + n0 + r + 8 * half) * CCH + ct * 16 + ln;
                out[o] = acc[r];
            }
        }
    }
}

// ---------------------------------------------------------------------------
extern "C" void kernel_launch(void* const* d_in, const int* in_sizes, int n_in,
                              void* d_out, int out_size, void* d_ws, size_t ws_size,
                              hipStream_t stream) {
    const float* x        = (const float*)d_in[0];
    const float* time_gra = (const float*)d_in[1];
    const float* w1       = (const float*)d_in[2];
    const float* b1       = (const float*)d_in[3];
    const float* w2       = (const float*)d_in[4];
    const float* b2       = (const float*)d_in[5];
    const float* emb_w    = (const float*)d_in[6];
    const float* emb_b    = (const float*)d_in[7];
    const float* gra_feat = (const float*)d_in[8];
    const float* A1r = (const float*)d_in[9],  *A1i = (const float*)d_in[10];
    const float* B1r = (const float*)d_in[11], *B1i = (const float*)d_in[12];
    const float* A2r = (const float*)d_in[13], *A2i = (const float*)d_in[14];
    const float* B2r = (const float*)d_in[15], *B2i = (const float*)d_in[16];
    float* out = (float*)d_out;

    char* ws = (char*)d_ws;
    __bf16* crci = (__bf16*)(ws + OFF_CRCI);
    __bf16* w1b  = (__bf16*)(ws + OFF_W1);
    __bf16* w2b  = (__bf16*)(ws + OFF_W2);
    __bf16* drdi = (__bf16*)(ws + OFF_DRDI);
    float*  fsec = (float*)(ws + OFF_F32);

    build_rfft_mat<<<(2 * CCH * FP + 255) / 256, 256, 0, stream>>>(crci);
    build_irfft_mat<<<(2 * FP * CCH + 255) / 256, 256, 0, stream>>>(drdi);
    build_w_mat<<<(2 * FP * FP + 255) / 256, 256, 0, stream>>>(w1, w1b);
    build_w_mat<<<(2 * FP * FP + 255) / 256, 256, 0, stream>>>(w2, w2b);
    build_gate<<<1, 128, 0, stream>>>(time_gra, emb_w, emb_b, gra_feat, b1, b2, fsec);

    dim3 grid(BB * (NN / 64));   // 1024 WGs, 4 wave32 each
    afno_main<<<grid, 128, SMEM_BYTES, stream>>>(
        x, crci, w1b, w2b, drdi, fsec,
        A1r, A1i, B1r, B1i, A2r, A2i, B2r, B2i, out);
}
